// DecoderMulti_45045617000948
// MI455X (gfx1250) — matrix operand order repaired and checked by reading, not compiled, not run on hardware
//
#include <hip/hip_runtime.h>

#define BATCH 128
#define SEQ   256
#define HIDN  512
#define EMBD  128
#define VOCAB 30000
#define ODIM  64
#define RDIM  32
#define INP   705      // HIDN + EMBD + 1 + ODIM
#define INPAD 736      // padded to multiple of 32
#define G3    1536     // 3*HIDN
#define ES_ROWS 128    // A rows staged in LDS per block
#define LDSTR 520      // padded LDS row stride in bf16 (1040 B -> conflict-free TR reads)

typedef __bf16 bf16;
typedef __attribute__((ext_vector_type(16))) __bf16 v16bf;
typedef __attribute__((ext_vector_type(8)))  __bf16 v8bf;
typedef __attribute__((ext_vector_type(8)))  float  v8f;

__device__ __forceinline__ bf16 f2bf(float f) {
  union { float f; unsigned u; } v; v.f = f;
  unsigned r = v.u + 0x7fffu + ((v.u >> 16) & 1u);
  unsigned short h = (unsigned short)(r >> 16);
  return __builtin_bit_cast(bf16, h);
}
__device__ __forceinline__ float sigmoidf(float x) { return 1.0f / (1.0f + __expf(-x)); }

// ---- CDNA5 async global->LDS copy (ASYNCcnt-tracked) ----
__device__ __forceinline__ void async_load_b128(unsigned lds_addr, const void* gaddr) {
  asm volatile("global_load_async_to_lds_b128 %0, %1, off"
               :: "v"(lds_addr), "v"(gaddr) : "memory");
}
__device__ __forceinline__ void wait_async0() {
  asm volatile("s_wait_asynccnt 0x0" ::: "memory");
}

// ---------------- conversion kernels ----------------
__global__ void cvt_kernel(const float* __restrict__ in, bf16* __restrict__ out, int n) {
  int i = blockIdx.x * blockDim.x + threadIdx.x;
  if (i < n) out[i] = f2bf(in[i]);
}
__global__ void cvt_pad_kernel(const float* __restrict__ in, bf16* __restrict__ out,
                               int rows, int K, int Kpad) {
  int i = blockIdx.x * blockDim.x + threadIdx.x;
  if (i >= rows * Kpad) return;
  int r = i / Kpad, k = i - r * Kpad;
  out[i] = (k < K) ? f2bf(in[(size_t)r * K + k]) : f2bf(0.0f);
}
// encoder_outputs (S,B,H) f32 -> (B,S,H) bf16
__global__ void cvt_enc_kernel(const float* __restrict__ in, bf16* __restrict__ out) {
  int i = blockIdx.x * blockDim.x + threadIdx.x;
  if (i >= SEQ * BATCH * HIDN) return;
  int s = i / (BATCH * HIDN);
  int rem = i - s * (BATCH * HIDN);
  int b = rem / HIDN, h = rem - b * HIDN;
  out[((size_t)b * SEQ + s) * HIDN + h] = f2bf(in[i]);
}
// attn_W (2H,H) -> Wt_hid[n][k]=attn_W[k][n], Wt_enc[n][k]=attn_W[H+k][n]
__global__ void cvt_attnW_kernel(const float* __restrict__ aw,
                                 bf16* __restrict__ wt_hid, bf16* __restrict__ wt_enc) {
  int i = blockIdx.x * blockDim.x + threadIdx.x;
  if (i >= HIDN * HIDN) return;
  int k = i / HIDN, n = i - k * HIDN;
  wt_hid[(size_t)n * HIDN + k] = f2bf(aw[(size_t)k * HIDN + n]);
  wt_enc[(size_t)n * HIDN + k] = f2bf(aw[(size_t)(HIDN + k) * HIDN + n]);
}

// ---------------- generic bf16 WMMA GEMM: C = A(MxK) * W^T (W is NxK) + bias ----------------
__global__ void gemm_bt_kernel(const bf16* __restrict__ A, int lda,
                               const bf16* __restrict__ W, int ldw,
                               const float* __restrict__ bias,
                               float* __restrict__ C, int ldc,
                               int K, int act) {
  const int lane = threadIdx.x;
  const int row  = lane & 15;
  const int hlf  = lane >> 4;
  const int m0 = blockIdx.y * 16, n0 = blockIdx.x * 16;
  const bf16* arow = A + (size_t)(m0 + row) * lda;
  const bf16* wrow = W + (size_t)(n0 + row) * ldw;
  v8f acc = {};
  for (int k0 = 0; k0 < K; k0 += 32) {
    v8bf a0 = *(const v8bf*)(arow + k0 + hlf * 8);
    v8bf a1 = *(const v8bf*)(arow + k0 + 16 + hlf * 8);
    v16bf a = __builtin_shufflevector(a0, a1, 0,1,2,3,4,5,6,7,8,9,10,11,12,13,14,15);
    v16bf b = __builtin_shufflevector(*(const v8bf*)(wrow + k0 + hlf * 8), *(const v8bf*)(wrow + k0 + 16 + hlf * 8), 0,1,2,3,4,5,6,7,8,9,10,11,12,13,14,15);
    acc = __builtin_amdgcn_wmma_f32_16x16x32_bf16(false, a, false, b, (short)0, acc, false, false);
  }
  const int cn = n0 + row;
  const float bia = bias ? bias[cn] : 0.0f;
#pragma unroll
  for (int r = 0; r < 8; r++) {
    float v = acc[r] + bia;
    if (act == 1) v = fmaxf(v, 0.0f);
    C[(size_t)(m0 + r + 8 * hlf) * ldc + cn] = v;
  }
}

// ---------------- attention energy + tanh + v_W dot -> score ----------------
// block: 256 thr (8 waves); stages 128 enc rows (LDS, async) shared by all waves;
// wave w handles n-tiles w*4 .. w*4+3; cross-wave score reduce via ds_add_f32.
__global__ void energy_score_kernel(const bf16* __restrict__ encb,   // (B,S,H)
                                    const bf16* __restrict__ wt_enc, // (H,H) n-major
                                    const float* __restrict__ hidp,  // (B,H) incl attn_b
                                    const float* __restrict__ vW,    // (H)
                                    float* __restrict__ score) {     // (B,S)
  extern __shared__ char smem[];
  bf16*  sA     = (bf16*)smem;                                   // 128 x LDSTR bf16
  float* sScore = (float*)(smem + (size_t)ES_ROWS * LDSTR * 2);  // 128 f32
  const int t  = threadIdx.x;
  const int s0 = blockIdx.x * ES_ROWS;
  const int b  = blockIdx.y;
  if (t < ES_ROWS) sScore[t] = 0.0f;
  // stage A strip: 128 rows x 512 bf16 in 16B chunks (8192 chunks / 256 thr = 32 each)
  const bf16* gA = encb + ((size_t)b * SEQ + s0) * HIDN;
#pragma unroll
  for (int it = 0; it < (ES_ROWS * HIDN / 8) / 256; ++it) {
    int chunk = t + it * 256;
    int r = chunk >> 6;           // 64 chunks per row
    int c = (chunk & 63) * 8;
    async_load_b128((unsigned)(size_t)(sA + r * LDSTR + c), gA + (size_t)r * HIDN + c);
  }
  wait_async0();
  __syncthreads();

  const int wave = t >> 5, lane = t & 31;
  const int row = lane & 15, hlf = lane >> 4;
  const float hpb = 0.0f; (void)hpb;
  float part[8][8] = {};
  for (int i = 0; i < 4; ++i) {
    const int n0 = (wave * 4 + i) * 16;
    const bf16* wrow = wt_enc + (size_t)(n0 + row) * HIDN;
    v8f acc[8] = {};
    for (int k0 = 0; k0 < HIDN; k0 += 32) {
      v16bf bm = __builtin_shufflevector(*(const v8bf*)(wrow + k0 + hlf * 8), *(const v8bf*)(wrow + k0 + 16 + hlf * 8), 0,1,2,3,4,5,6,7,8,9,10,11,12,13,14,15);
#pragma unroll
      for (int mt = 0; mt < 8; ++mt) {
        const bf16* arow = sA + (size_t)(mt * 16 + row) * LDSTR;
        v8bf a0 = *(const v8bf*)(arow + k0 + hlf * 8);
        v8bf a1 = *(const v8bf*)(arow + k0 + 16 + hlf * 8);
        v16bf a = __builtin_shufflevector(a0, a1, 0,1,2,3,4,5,6,7,8,9,10,11,12,13,14,15);
        acc[mt] = __builtin_amdgcn_wmma_f32_16x16x32_bf16(false, a, false, bm, (short)0, acc[mt], false, false);
      }
    }
    const int n = n0 + row;
    const float hp = hidp[(size_t)b * HIDN + n];
    const float vw = vW[n];
#pragma unroll
    for (int mt = 0; mt < 8; ++mt)
#pragma unroll
      for (int r = 0; r < 8; ++r)
        part[mt][r] += tanhf(acc[mt][r] + hp) * vw;
  }
  // reduce each partial over the 16 lanes of its half (the N dimension), then LDS-atomic
#pragma unroll
  for (int mt = 0; mt < 8; ++mt)
#pragma unroll
    for (int r = 0; r < 8; ++r) {
      float p = part[mt][r];
      for (int off = 1; off < 16; off <<= 1) p += __shfl_xor(p, off, 32);
      if (row == 0) atomicAdd(&sScore[mt * 16 + r + 8 * hlf], p);
    }
  __syncthreads();
  if (t < ES_ROWS) score[(size_t)b * SEQ + s0 + t] = sScore[t];
}

// ---------------- masked softmax over S (block per batch) ----------------
__global__ void softmax_kernel(const float* __restrict__ score, const int* __restrict__ mask,
                               float* __restrict__ attw) {
  __shared__ float red[256];
  const int b = blockIdx.x, t = threadIdx.x;
  float v = (mask[(size_t)b * SEQ + t] == 0) ? -1e10f : score[(size_t)b * SEQ + t];
  red[t] = v; __syncthreads();
  for (int o = 128; o > 0; o >>= 1) { if (t < o) red[t] = fmaxf(red[t], red[t + o]); __syncthreads(); }
  const float m = red[0]; __syncthreads();
  float e = __expf(v - m);
  red[t] = e; __syncthreads();
  for (int o = 128; o > 0; o >>= 1) { if (t < o) red[t] += red[t + o]; __syncthreads(); }
  attw[(size_t)b * SEQ + t] = e / red[0];
}

// ---------------- weighted context + build padded bf16 rnn_in row ----------------
__global__ void weighted_rnnin_kernel(const bf16* __restrict__ encb, const float* __restrict__ attw,
                                      const int* __restrict__ input_id, const float* __restrict__ input_rate,
                                      const float* __restrict__ onlf, const float* __restrict__ emb,
                                      bf16* __restrict__ rnn_in) {
  __shared__ float aS[SEQ];
  const int b = blockIdx.x, t = threadIdx.x;
  aS[t] = attw[(size_t)b * SEQ + t];
  __syncthreads();
  bf16* rrow = rnn_in + (size_t)b * INPAD;
  const bf16* eb = encb + (size_t)b * SEQ * HIDN;
#pragma unroll
  for (int j = t; j < HIDN; j += 256) {
    float acc = 0.f;
    for (int s = 0; s < SEQ; s++) acc += aS[s] * (float)eb[(size_t)s * HIDN + j];
    rrow[j] = f2bf(acc);
  }
  const int id = input_id[b];
  if (t < EMBD)                 rrow[HIDN + t] = f2bf(emb[(size_t)id * EMBD + t]);
  if (t == EMBD)                rrow[HIDN + EMBD] = f2bf(input_rate[b]);
  if (t > EMBD && t <= EMBD + ODIM) rrow[HIDN + EMBD + t - EMBD] = f2bf(onlf[(size_t)b * ODIM + (t - EMBD - 1)]);
  if (t > EMBD + ODIM && t < EMBD + ODIM + 1 + (INPAD - INP)) rrow[INP + (t - EMBD - ODIM - 1)] = f2bf(0.0f);
}

// ---------------- GRU gate combine ----------------
__global__ void gru_kernel(const float* __restrict__ gi, const float* __restrict__ gh,
                           const float* __restrict__ h, float* __restrict__ hnew,
                           bf16* __restrict__ hnew_bf) {
  int i = blockIdx.x * blockDim.x + threadIdx.x;
  if (i >= BATCH * HIDN) return;
  int b = i / HIDN, j = i - b * HIDN;
  const float* gib = gi + (size_t)b * G3;
  const float* ghb = gh + (size_t)b * G3;
  float r = sigmoidf(gib[j] + ghb[j]);
  float z = sigmoidf(gib[HIDN + j] + ghb[HIDN + j]);
  float n = tanhf(gib[2 * HIDN + j] + r * ghb[2 * HIDN + j]);
  float hv = (1.0f - z) * n + z * h[i];
  hnew[i] = hv;
  hnew_bf[i] = f2bf(hv);
}

// ---------------- logits: 8 waves/block, h_new staged in LDS, fc_id_W streamed once ----------------
__global__ void logits_kernel(const bf16* __restrict__ hnew_bf, const bf16* __restrict__ fcW,
                              const float* __restrict__ fcb, float* __restrict__ logits) {
  extern __shared__ char smem[];
  bf16* sA = (bf16*)smem;                 // 128 x LDSTR bf16
  const int t = threadIdx.x;
#pragma unroll
  for (int it = 0; it < (BATCH * HIDN / 8) / 256; ++it) {
    int chunk = t + it * 256;
    int r = chunk >> 6;
    int c = (chunk & 63) * 8;
    async_load_b128((unsigned)(size_t)(sA + r * LDSTR + c), hnew_bf + (size_t)r * HIDN + c);
  }
  wait_async0();
  __syncthreads();

  const int wave = t >> 5, lane = t & 31;
  const int nt = blockIdx.x * 8 + wave;
  if (nt >= VOCAB / 16) return;
  const int n0 = nt * 16;
  const int row = lane & 15, hlf = lane >> 4;
  const bf16* wrow = fcW + (size_t)(n0 + row) * HIDN;
  v8f acc[8] = {};
  for (int k0 = 0; k0 < HIDN; k0 += 32) {
    v16bf bm = __builtin_shufflevector(*(const v8bf*)(wrow + k0 + hlf * 8), *(const v8bf*)(wrow + k0 + 16 + hlf * 8), 0,1,2,3,4,5,6,7,8,9,10,11,12,13,14,15);
#pragma unroll
    for (int mt = 0; mt < 8; mt++) {
      const bf16* arow = sA + (size_t)(mt * 16 + row) * LDSTR;
      v8bf a0 = *(const v8bf*)(arow + k0 + hlf * 8);
      v8bf a1 = *(const v8bf*)(arow + k0 + 16 + hlf * 8);
      v16bf a = __builtin_shufflevector(a0, a1, 0,1,2,3,4,5,6,7,8,9,10,11,12,13,14,15);
      acc[mt] = __builtin_amdgcn_wmma_f32_16x16x32_bf16(false, a, false, bm, (short)0, acc[mt], false, false);
    }
  }
  const int cn = n0 + row;
  const float bia = fcb[cn];
#pragma unroll
  for (int mt = 0; mt < 8; mt++)
#pragma unroll
    for (int r = 0; r < 8; r++)
      logits[(size_t)(mt * 16 + r + 8 * hlf) * VOCAB + cn] = acc[mt][r] + bia;
}

// ---------------- constrained log-softmax + argmax over VOCAB ----------------
__global__ void csoftmax_kernel(float* __restrict__ pid, const float* __restrict__ cvec,
                                int* __restrict__ maxid) {
  __shared__ float red[256];
  __shared__ float bv[256];
  __shared__ int   bi[256];
  const int b = blockIdx.x, t = threadIdx.x;
  float* lg = pid + (size_t)b * VOCAB;
  const float* cv = cvec + (size_t)b * VOCAB;
  float m = -3.4e38f;
  for (int i = t; i < VOCAB; i += 256) m = fmaxf(m, lg[i]);
  red[t] = m; __syncthreads();
  for (int o = 128; o > 0; o >>= 1) { if (t < o) red[t] = fmaxf(red[t], red[t + o]); __syncthreads(); }
  m = red[0]; __syncthreads();
  float s = 0.f;
  for (int i = t; i < VOCAB; i += 256) s += __expf(lg[i] - m) * cv[i];
  red[t] = s; __syncthreads();
  for (int o = 128; o > 0; o >>= 1) { if (t < o) red[t] += red[t + o]; __syncthreads(); }
  s = red[0]; __syncthreads();
  float best = -1.0f; int besti = VOCAB;
  const float ninf = -__builtin_inff();
  for (int i = t; i < VOCAB; i += 256) {
    float c = cv[i];
    float xe = __expf(lg[i] - m) * c;
    lg[i] = (c > 0.f) ? __logf(xe / s) : ninf;
    if (c > 0.f && (xe > best || (xe == best && i < besti))) { best = xe; besti = i; }
  }
  bv[t] = best; bi[t] = besti; __syncthreads();
  for (int o = 128; o > 0; o >>= 1) {
    if (t < o) {
      if (bv[t + o] > bv[t] || (bv[t + o] == bv[t] && bi[t + o] < bi[t])) { bv[t] = bv[t + o]; bi[t] = bi[t + o]; }
    }
    __syncthreads();
  }
  if (t == 0) maxid[b] = bi[0];
}

// ---------------- build rate-head A matrix: [emb[max_id] | h_new] bf16 ----------------
__global__ void rateA_kernel(const int* __restrict__ maxid, const float* __restrict__ emb,
                             const bf16* __restrict__ hnew_bf, bf16* __restrict__ rateA) {
  const int b = blockIdx.x, t = threadIdx.x;
  bf16* rr = rateA + (size_t)b * (EMBD + HIDN);
  const int id = maxid[b];
  if (t < EMBD) rr[t] = f2bf(emb[(size_t)id * EMBD + t]);
  for (int j = t; j < HIDN; j += 256) rr[EMBD + j] = hnew_bf[(size_t)b * HIDN + j];
}

// ---------------- final rate head ----------------
__global__ void rate_kernel(const float* __restrict__ rate_in, const float* __restrict__ ridf,
                            const float* __restrict__ rate_W, const float* __restrict__ rate_b,
                            float* __restrict__ prate) {
  const int b = threadIdx.x;
  if (b >= BATCH) return;
  float acc = rate_b[0];
  for (int k = 0; k < HIDN; k++) acc += rate_in[(size_t)b * HIDN + k] * rate_W[k];
  for (int k = 0; k < RDIM; k++) acc += ridf[(size_t)b * RDIM + k] * rate_W[HIDN + k];
  prate[b] = sigmoidf(acc);
}

// =====================================================================
extern "C" void kernel_launch(void* const* d_in, const int* in_sizes, int n_in,
                              void* d_out, int out_size, void* d_ws, size_t ws_size,
                              hipStream_t stream) {
  const int*   input_id   = (const int*)  d_in[0];
  const float* input_rate = (const float*)d_in[1];
  const float* hidden     = (const float*)d_in[2];
  const float* enc        = (const float*)d_in[3];
  const int*   attn_mask  = (const int*)  d_in[4];
  const float* cvec       = (const float*)d_in[5];
  const float* onlf       = (const float*)d_in[6];
  const float* ridf       = (const float*)d_in[7];
  const float* emb        = (const float*)d_in[8];
  const float* attn_b     = (const float*)d_in[10];
  const float* v_W        = (const float*)d_in[11];
  const float* b_ih       = (const float*)d_in[13];
  const float* b_hh       = (const float*)d_in[15];
  const float* fc_id_b    = (const float*)d_in[17];
  const float* tan_b      = (const float*)d_in[19];
  const float* rate_W     = (const float*)d_in[20];
  const float* rate_b     = (const float*)d_in[21];

  float* out = (float*)d_out;
  float* pid   = out;                                  // (B, VOCAB)
  float* prate = out + (size_t)BATCH * VOCAB;          // (B, 1)
  float* hnew  = prate + BATCH;                        // (1, B, HIDN)

  char* wsb = (char*)d_ws;
  size_t cur = 0;
  auto alloc = [&](size_t bytes) -> char* {
    char* p = wsb + cur;
    cur += (bytes + 255) & ~(size_t)255;
    return p;
  };
  bf16* enc_bf   = (bf16*)alloc((size_t)BATCH * SEQ * HIDN * 2);
  bf16* wt_hid   = (bf16*)alloc((size_t)HIDN * HIDN * 2);
  bf16* wt_enc   = (bf16*)alloc((size_t)HIDN * HIDN * 2);
  bf16* fcW_bf   = (bf16*)alloc((size_t)VOCAB * HIDN * 2);
  bf16* wih_bf   = (bf16*)alloc((size_t)G3 * INPAD * 2);
  bf16* whh_bf   = (bf16*)alloc((size_t)G3 * HIDN * 2);
  bf16* tanW_bf  = (bf16*)alloc((size_t)HIDN * (EMBD + HIDN) * 2);
  bf16* h_bf     = (bf16*)alloc((size_t)BATCH * HIDN * 2);
  float* hidp    = (float*)alloc((size_t)BATCH * HIDN * 4);
  float* score   = (float*)alloc((size_t)BATCH * SEQ * 4);
  float* attw    = (float*)alloc((size_t)BATCH * SEQ * 4);
  bf16* rnn_in   = (bf16*)alloc((size_t)BATCH * INPAD * 2);
  float* gi      = (float*)alloc((size_t)BATCH * G3 * 4);
  float* gh      = (float*)alloc((size_t)BATCH * G3 * 4);
  bf16* hnew_bf  = (bf16*)alloc((size_t)BATCH * HIDN * 2);
  int*  maxid    = (int*)alloc((size_t)BATCH * 4);
  bf16* rateA    = (bf16*)alloc((size_t)BATCH * (EMBD + HIDN) * 2);
  float* rate_in = (float*)alloc((size_t)BATCH * HIDN * 4);
  (void)ws_size; (void)in_sizes; (void)n_in; (void)out_size;

  const int T = 256;
  const size_t LDS_A = (size_t)ES_ROWS * LDSTR * 2;            // 133,120 B
  const size_t LDS_ES = LDS_A + ES_ROWS * sizeof(float);       // + score accum

  // ---- precision conversion / layout passes ----
  cvt_enc_kernel<<<(SEQ * BATCH * HIDN + T - 1) / T, T, 0, stream>>>(enc, enc_bf);
  cvt_attnW_kernel<<<(HIDN * HIDN + T - 1) / T, T, 0, stream>>>((const float*)d_in[9], wt_hid, wt_enc);
  cvt_kernel<<<(BATCH * HIDN + T - 1) / T, T, 0, stream>>>(hidden, h_bf, BATCH * HIDN);
  cvt_kernel<<<(VOCAB * HIDN + T - 1) / T, T, 0, stream>>>((const float*)d_in[16], fcW_bf, VOCAB * HIDN);
  cvt_kernel<<<(G3 * HIDN + T - 1) / T, T, 0, stream>>>((const float*)d_in[14], whh_bf, G3 * HIDN);
  cvt_kernel<<<(HIDN * (EMBD + HIDN) + T - 1) / T, T, 0, stream>>>((const float*)d_in[18], tanW_bf, HIDN * (EMBD + HIDN));
  cvt_pad_kernel<<<(G3 * INPAD + T - 1) / T, T, 0, stream>>>((const float*)d_in[12], wih_bf, G3, INP, INPAD);

  // ---- hidp = h @ attn_W[:H] + attn_b ----
  gemm_bt_kernel<<<dim3(HIDN / 16, BATCH / 16), 32, 0, stream>>>(
      h_bf, HIDN, wt_hid, HIDN, attn_b, hidp, HIDN, HIDN, 0);

  // ---- attention scores: LDS-staged A (async), 8 waves/block ----
  energy_score_kernel<<<dim3(SEQ / ES_ROWS, BATCH), 256, LDS_ES, stream>>>(
      enc_bf, wt_enc, hidp, v_W, score);

  // ---- masked softmax over S ----
  softmax_kernel<<<BATCH, SEQ, 0, stream>>>(score, attn_mask, attw);

  // ---- context vector + rnn_in assembly ----
  weighted_rnnin_kernel<<<BATCH, T, 0, stream>>>(enc_bf, attw, input_id, input_rate, onlf, emb, rnn_in);

  // ---- GRU gate GEMMs ----
  gemm_bt_kernel<<<dim3(G3 / 16, BATCH / 16), 32, 0, stream>>>(
      rnn_in, INPAD, wih_bf, INPAD, b_ih, gi, G3, INPAD, 0);
  gemm_bt_kernel<<<dim3(G3 / 16, BATCH / 16), 32, 0, stream>>>(
      h_bf, HIDN, whh_bf, HIDN, b_hh, gh, G3, HIDN, 0);

  // ---- GRU combine -> h_new (output 3) ----
  gru_kernel<<<(BATCH * HIDN + T - 1) / T, T, 0, stream>>>(gi, gh, hidden, hnew, hnew_bf);

  // ---- logits: h_new staged in LDS (async), fc_id_W streamed once ----
  logits_kernel<<<(VOCAB / 16 + 7) / 8, 256, LDS_A, stream>>>(hnew_bf, fcW_bf, fc_id_b, pid);

  // ---- constrained log-softmax + argmax (in-place on pid, output 1) ----
  csoftmax_kernel<<<BATCH, T, 0, stream>>>(pid, cvec, maxid);

  // ---- rate head ----
  rateA_kernel<<<BATCH, T, 0, stream>>>(maxid, emb, hnew_bf, rateA);
  gemm_bt_kernel<<<dim3(HIDN / 16, BATCH / 16), 32, 0, stream>>>(
      rateA, EMBD + HIDN, tanW_bf, EMBD + HIDN, tan_b, rate_in, HIDN, EMBD + HIDN, 1);
  rate_kernel<<<1, BATCH, 0, stream>>>(rate_in, ridf, rate_W, rate_b, prate);
}
